// PointNetSetAbstractionMsg_9251359555686
// MI455X (gfx1250) — matrix operand order repaired and checked by reading, not compiled, not run on hardware
//
#include <hip/hip_runtime.h>
#include <hip/hip_bf16.h>

typedef _Float16 h16;
typedef __attribute__((ext_vector_type(16))) _Float16 v16h;
typedef __attribute__((ext_vector_type(8)))  _Float16 v8h;
typedef __attribute__((ext_vector_type(8)))  float    v8f;

// Problem constants (from reference)
#define cB   16
#define cN   4096
#define cS   1024
#define cIN  32
#define TOTCH 320   // 64 + 128 + 128

// ---------------- workspace layout (bytes) ----------------
#define OFF_SAMPLED 0u                      // cB*cS int           = 64 KB
#define OFF_NEWXYZ  (65536u)                // cB*cS*3 f32         = 192 KB
#define OFF_GIDX    (262144u)               // cB*cS*64 int        = 4 MB
#define OFF_W16     (4456448u)              // <= 96*128 f16       = 24 KB (pad 32K)
#define OFF_STATS   (4489216u)              // 4*128 f32 (sum|sq|scale|shift)
#define OFF_BUFA    (4493312u)              // 1M rows * 96 f16    = 192 MB
#define OFF_BUFB    (205819904u)            // 1M rows * 64 f16    = 128 MB

// =========================================================
// 1) Farthest point sampling: one workgroup per batch.
//    Points + running distances live in registers (16/thread).
// =========================================================
__global__ void __launch_bounds__(256)
fps_kernel(const float* __restrict__ xyz, int* __restrict__ sampled)
{
    const int b = blockIdx.x, tid = threadIdx.x;
    const int PPT = cN / 256; // 16 points per thread
    float px[PPT], py[PPT], pz[PPT], dist[PPT];
#pragma unroll
    for (int i = 0; i < PPT; ++i) {
        int p = tid + i * 256;
        size_t base = ((size_t)b * cN + p) * 3;
        px[i] = xyz[base + 0];
        py[i] = xyz[base + 1];
        pz[i] = xyz[base + 2];
        dist[i] = 1e10f;
    }
    __shared__ float rv[256];
    __shared__ int   ri[256];
    __shared__ float cx, cy, cz;
    int farthest = 0;
    for (int s = 0; s < cS; ++s) {
        if (tid == 0) {
            sampled[b * cS + s] = farthest;
            size_t base = ((size_t)b * cN + farthest) * 3;
            cx = xyz[base + 0]; cy = xyz[base + 1]; cz = xyz[base + 2];
        }
        __syncthreads();
        float bv = -1.0f; int bi = 0;
        float lcx = cx, lcy = cy, lcz = cz;
#pragma unroll
        for (int i = 0; i < PPT; ++i) {
            float dx = px[i] - lcx, dy = py[i] - lcy, dz = pz[i] - lcz;
            float d = dx * dx + dy * dy + dz * dz;
            dist[i] = fminf(dist[i], d);
            if (dist[i] > bv) { bv = dist[i]; bi = tid + i * 256; }
        }
        rv[tid] = bv; ri[tid] = bi;
        __syncthreads();
        for (int st = 128; st > 0; st >>= 1) {
            if (tid < st) {
                if (rv[tid + st] > rv[tid] ||
                    (rv[tid + st] == rv[tid] && ri[tid + st] < ri[tid])) {
                    rv[tid] = rv[tid + st]; ri[tid] = ri[tid + st];
                }
            }
            __syncthreads();
        }
        farthest = ri[0];
        __syncthreads();
    }
}

// =========================================================
// 2) Gather sampled centers; also emit transposed new_xyz output.
// =========================================================
__global__ void __launch_bounds__(256)
gather_newxyz_kernel(const float* __restrict__ xyz, const int* __restrict__ sampled,
                     float* __restrict__ newxyz, float* __restrict__ out)
{
    int i = blockIdx.x * 256 + threadIdx.x;
    if (i >= cB * cS) return;
    int b = i / cS, s = i - b * cS;
    int p = sampled[i];
#pragma unroll
    for (int c = 0; c < 3; ++c) {
        float v = xyz[((size_t)b * cN + p) * 3 + c];
        newxyz[(size_t)i * 3 + c] = v;
        out[((size_t)b * 3 + c) * cS + s] = v;   // new_xyz^T (B,3,S)
    }
}

__global__ void __launch_bounds__(256)
init_out_kernel(float* __restrict__ out, int n)
{
    int i = blockIdx.x * 256 + threadIdx.x;
    if (i < n) out[i] = 0.0f;
}

// =========================================================
// 3) Ball query: one wave32 per query; ballot extracts first-K
//    hits in ascending index order (matches sort-truncate ref).
// =========================================================
__global__ void __launch_bounds__(256)
query_ball_kernel(const float* __restrict__ xyz, const float* __restrict__ newxyz,
                  int* __restrict__ gidx, float r2, int Kg)
{
    int lane = threadIdx.x & 31;
    int q = (blockIdx.x * 256 + threadIdx.x) >> 5;
    if (q >= cB * cS) return;
    int b = q >> 10;   // q / cS
    float cx = newxyz[(size_t)q * 3 + 0];
    float cy = newxyz[(size_t)q * 3 + 1];
    float cz = newxyz[(size_t)q * 3 + 2];
    const float* base = xyz + (size_t)b * cN * 3;
    int cnt = 0, first = 0;
    for (int p0 = 0; p0 < cN && cnt < Kg; p0 += 32) {
        int p = p0 + lane;
        float dx = base[p * 3 + 0] - cx;
        float dy = base[p * 3 + 1] - cy;
        float dz = base[p * 3 + 2] - cz;
        float d2 = dx * dx + dy * dy + dz * dz;
        unsigned mask = (unsigned)__ballot(d2 <= r2);
        if (lane == 0) {
            while (mask && cnt < Kg) {
                int t = __ffs(mask) - 1;
                if (cnt == 0) first = p0 + t;
                gidx[(size_t)q * Kg + cnt] = p0 + t;
                ++cnt;
                mask &= mask - 1;
            }
        }
        cnt = __shfl(cnt, 0);
    }
    if (lane == 0) {
        for (int j = cnt; j < Kg; ++j) gidx[(size_t)q * Kg + j] = first;
    }
}

// =========================================================
// 4) Build layer-0 input rows: [points(32) | xyz-center(3) | pad->64] as f16
// =========================================================
__global__ void __launch_bounds__(256)
build_h0_kernel(const float* __restrict__ xyz, const float* __restrict__ pts,
                const int* __restrict__ gidx, const float* __restrict__ newxyz,
                h16* __restrict__ A, int kShift, int rows)
{
    int r = blockIdx.x * 256 + threadIdx.x;
    if (r >= rows) return;
    int g = r >> kShift;            // (b*S + s)
    int b = g >> 10;                // g / cS
    int p = gidx[r];
    if (p > cN - 1) p = cN - 1;
    const float* src = pts + ((size_t)b * cN + p) * cIN;
    h16 tmp[64];
#pragma unroll
    for (int c = 0; c < cIN; ++c) tmp[c] = (h16)src[c];
    const float* pxyz = xyz + ((size_t)b * cN + p) * 3;
    const float* cen  = newxyz + (size_t)g * 3;
    tmp[32] = (h16)(pxyz[0] - cen[0]);
    tmp[33] = (h16)(pxyz[1] - cen[1]);
    tmp[34] = (h16)(pxyz[2] - cen[2]);
#pragma unroll
    for (int c = 35; c < 64; ++c) tmp[c] = (h16)0.0f;
    v8h* dst = (v8h*)(A + (size_t)r * 64);    // 128B-aligned row
#pragma unroll
    for (int c = 0; c < 8; ++c) dst[c] = *(v8h*)&tmp[c * 8];
}

// =========================================================
// 5) Weight convert: W(out,Cin) f32 -> W^T padded, pre-swizzled into
//    WMMA B-fragment order: [kTile][colTile][lane(32)][i(16)] halves,
//    where K = kTile*32 + (lane>>4)*16 + i, o = colTile*16 + (lane&15).
//    GEMM then LDS-stages with plain copies and reads v16h per lane.
// =========================================================
__global__ void __launch_bounds__(256)
conv_w_kernel(const float* __restrict__ Wsrc, h16* __restrict__ Wd,
              int Cin, int Kp, int Cout)
{
    int i = blockIdx.x * 256 + threadIdx.x;
    if (i >= Kp * Cout) return;
    int colTiles = Cout >> 4;
    int tileIdx = i >> 9;              // 512 halves per (kTile,colTile) fragment
    int within  = i & 511;
    int lane = within >> 4, ii = within & 15;
    int kTile = tileIdx / colTiles;
    int colTile = tileIdx - kTile * colTiles;
    int K = kTile * 32 + ((lane >> 4) << 4) + ii;
    int o = colTile * 16 + (lane & 15);
    Wd[i] = (K < Cin) ? (h16)Wsrc[o * Cin + K] : (h16)0.0f;
}

__global__ void zero_stats_kernel(float* __restrict__ stats)
{
    stats[threadIdx.x] = 0.0f;   // <<<1,256>>> clears sum[128]+sumsq[128]
}

__global__ void scale_shift_kernel(float* __restrict__ stats,
                                   const float* __restrict__ gamma,
                                   const float* __restrict__ beta,
                                   int Cout, float invCnt)
{
    int o = threadIdx.x;
    if (o >= Cout) return;
    float mean = stats[o] * invCnt;
    float var  = stats[128 + o] * invCnt - mean * mean;
    float sc   = gamma[o] * rsqrtf(var + 1e-5f);
    stats[256 + o] = sc;
    stats[384 + o] = beta[o] - mean * sc;
}

// =========================================================
// 6) WMMA GEMM: out[r,o] = sum_c A[r,c] * W^T[c,o]
//    Each wave: 64 rows x 16 cols macro-tile (4 accumulators),
//    B fragment reused 4x, loaded as one v16h from LDS.
//    mode 0: per-channel sum / sumsq   (stats pass)
//    mode 1: BN+ReLU, store f16 -> next layer input
//    mode 2: BN+ReLU, fused max over K -> new_points^T
// =========================================================
__global__ void __launch_bounds__(256)
gemm_wmma_kernel(const h16* __restrict__ A, const h16* __restrict__ Wg,
                 int rows, int Kp, int Cout, int mode,
                 float* __restrict__ stats,
                 h16* __restrict__ Bout,
                 float* __restrict__ outBase, int kShift, int chOff)
{
    __shared__ __align__(32) h16 Wl[96 * 128];   // whole swizzled W^T, <= 24 KB
    __shared__ float lsum[128], lsq[128];
    int tid = threadIdx.x;
    if (mode == 0 && tid < 128) { lsum[tid] = 0.0f; lsq[tid] = 0.0f; }
    {   // vectorized LDS stage: already in fragment order
        int n4 = (Kp * Cout) >> 3;               // uint4 = 8 halves
        const uint4* src = (const uint4*)Wg;
        uint4* dst = (uint4*)Wl;
        for (int i = tid; i < n4; i += 256) dst[i] = src[i];
    }
    __syncthreads();

    int lane = tid & 31, wid = tid >> 5;
    int colTiles = Cout >> 4;
    int rowBlocks = rows >> 6;                   // 64 rows per wave task
    int task = blockIdx.x * 8 + wid;
    bool active = task < rowBlocks * colTiles;
    int rowBlk = active ? (task / colTiles) : 0;
    int colTile = task - rowBlk * colTiles;
    int nOut = colTile * 16 + (lane & 15);

    v8f acc[4] = {{}, {}, {}, {}};
    if (active) {
        int kb = (lane >> 4) * 16;               // A per-lane K sub-block
        const h16* Ab = A + ((size_t)rowBlk * 64 + (lane & 15)) * Kp;
        const h16* Wb = Wl + (size_t)colTile * 512 + lane * 16;
        int kSteps = Kp >> 5;
        for (int ks = 0; ks < kSteps; ++ks) {
            int k0 = ks << 5;
            v16h bv = *(const v16h*)(Wb + (size_t)ks * colTiles * 512);
#pragma unroll
            for (int r = 0; r < 4; ++r) {
                const h16* p = Ab + (size_t)(r * 16) * Kp + k0 + kb;
                v8h lo = *(const v8h*)p;
                v8h hi = *(const v8h*)(p + 8);
                v16h av;
#pragma unroll
                for (int i = 0; i < 8; ++i) { av[i] = lo[i]; av[i + 8] = hi[i]; }
                acc[r] = __builtin_amdgcn_wmma_f32_16x16x32_f16(
                    false, av, false, bv, (short)0, acc[r], false, false);
            }
        }
    }

    if (mode == 0) {
        if (active) {
            float s = 0.0f, q = 0.0f;
#pragma unroll
            for (int r = 0; r < 4; ++r)
#pragma unroll
                for (int j = 0; j < 8; ++j) { float v = acc[r][j]; s += v; q += v * v; }
            atomicAdd(&lsum[nOut], s);
            atomicAdd(&lsq[nOut], q);
        }
        __syncthreads();
        if (tid < Cout) {
            atomicAdd(&stats[tid], lsum[tid]);
            atomicAdd(&stats[128 + tid], lsq[tid]);
        }
    } else if (active) {
        float sc = stats[256 + nOut], sh = stats[384 + nOut];
        if (mode == 1) {
            int roff = (lane >= 16) ? 8 : 0;
#pragma unroll
            for (int r = 0; r < 4; ++r) {
                int rbase = rowBlk * 64 + r * 16 + roff;
#pragma unroll
                for (int j = 0; j < 8; ++j) {
                    float y = fmaxf(acc[r][j] * sc + sh, 0.0f);
                    Bout[(size_t)(rbase + j) * Cout + nOut] = (h16)y;  // next Kp == Cout
                }
            }
        } else {
#pragma unroll
            for (int r = 0; r < 4; ++r) {
                float m = 0.0f;
#pragma unroll
                for (int j = 0; j < 8; ++j)
                    m = fmaxf(m, fmaxf(acc[r][j] * sc + sh, 0.0f));
                // 16 rows of a subtile belong to one (b,s) group (K >= 16, aligned)
                int g = (rowBlk * 64 + r * 16) >> kShift;
                int b = g >> 10, s = g & (cS - 1);
                unsigned* dst = (unsigned*)&outBase[((size_t)b * TOTCH + chOff + nOut) * cS + s];
                atomicMax(dst, __float_as_uint(m));  // monotone: all values >= 0
            }
        }
    }
}

// =========================================================
// Host orchestration
// =========================================================
extern "C" void kernel_launch(void* const* d_in, const int* in_sizes, int n_in,
                              void* d_out, int out_size, void* d_ws, size_t ws_size,
                              hipStream_t stream)
{
    const float* xyz = (const float*)d_in[0];
    const float* pts = (const float*)d_in[1];
    float* out = (float*)d_out;

    char* ws = (char*)d_ws;
    int*   sampled = (int*)(ws + OFF_SAMPLED);
    float* newxyz  = (float*)(ws + OFF_NEWXYZ);
    int*   gidx    = (int*)(ws + OFF_GIDX);
    h16*   Wf      = (h16*)(ws + OFF_W16);
    float* stats   = (float*)(ws + OFF_STATS);
    h16*   bufA    = (h16*)(ws + OFF_BUFA);
    h16*   bufB    = (h16*)(ws + OFF_BUFB);

    // FPS + centers + output init
    fps_kernel<<<cB, 256, 0, stream>>>(xyz, sampled);
    gather_newxyz_kernel<<<(cB * cS + 255) / 256, 256, 0, stream>>>(xyz, sampled, newxyz, out);
    const int nptsOut = cB * TOTCH * cS;
    init_out_kernel<<<(nptsOut + 255) / 256, 256, 0, stream>>>(out + cB * 3 * cS, nptsOut);

    const int   mlp[3][3] = {{32, 32, 64}, {64, 64, 128}, {64, 96, 128}};
    const int   KS[3]     = {16, 32, 64};
    const int   KSH[3]    = {4, 5, 6};         // log2(K)
    const float RAD[3]    = {0.1f, 0.2f, 0.4f};
    const int   chOff[3]  = {0, 64, 192};

    for (int i = 0; i < 3; ++i) {
        int Kg = KS[i], kShift = KSH[i];
        int rows = cB * cS * Kg;
        query_ball_kernel<<<(cB * cS * 32 + 255) / 256, 256, 0, stream>>>(
            xyz, newxyz, gidx, RAD[i] * RAD[i], Kg);
        build_h0_kernel<<<(rows + 255) / 256, 256, 0, stream>>>(
            xyz, pts, gidx, newxyz, bufA, kShift, rows);

        for (int l = 0; l < 3; ++l) {
            int Cin  = (l == 0) ? (cIN + 3) : mlp[i][l - 1];
            int Kp   = (l == 0) ? 64 : mlp[i][l - 1];  // padded contraction dim
            int Cout = mlp[i][l];
            int pidx = 2 + (i * 3 + l) * 4;
            const float* Wsrc  = (const float*)d_in[pidx + 0];
            const float* gamma = (const float*)d_in[pidx + 2];
            const float* beta  = (const float*)d_in[pidx + 3];

            conv_w_kernel<<<(Kp * Cout + 255) / 256, 256, 0, stream>>>(Wsrc, Wf, Cin, Kp, Cout);
            zero_stats_kernel<<<1, 256, 0, stream>>>(stats);

            const h16* Ain = (l == 1) ? bufB : bufA;   // ping-pong: A -> B -> A -> out
            h16* Bo        = (l == 0) ? bufB : bufA;
            int waveTasks = (rows >> 6) * (Cout >> 4);
            int blocks = (waveTasks + 7) / 8;

            // Pass 1: stats (sum / sumsq per channel)
            gemm_wmma_kernel<<<blocks, 256, 0, stream>>>(
                Ain, Wf, rows, Kp, Cout, 0, stats, nullptr, nullptr, kShift, 0);
            scale_shift_kernel<<<1, 128, 0, stream>>>(stats, gamma, beta, Cout, 1.0f / (float)rows);
            // Pass 2: apply BN+ReLU (+ fused max-over-K on last layer)
            int mode = (l == 2) ? 2 : 1;
            gemm_wmma_kernel<<<blocks, 256, 0, stream>>>(
                Ain, Wf, rows, Kp, Cout, mode, stats, Bo, out + cB * 3 * cS, kShift, chOff[i]);
        }
    }
}
